// AttentionModel_39848706573388
// MI455X (gfx1250) — matrix operand order repaired and checked by reading, not compiled, not run on hardware
//
#include <hip/hip_runtime.h>
#include <stdint.h>

// Problem constants (from reference): B=2, S=2048, H=8, E=64, all f16.
#define B_ 2
#define S_ 2048
#define H_ 8
#define E_ 64
#define BH_ (B_ * H_)
#define CHUNK 64                 // KV columns staged per TDM transfer
#define NCHUNK (S_ / CHUNK)      // 32

typedef __attribute__((ext_vector_type(16))) _Float16     v16h;
typedef __attribute__((ext_vector_type(8)))  _Float16     v8h;
typedef __attribute__((ext_vector_type(8)))  float        v8f;
typedef __attribute__((ext_vector_type(4)))  unsigned int v4u;
typedef __attribute__((ext_vector_type(8)))  int          v8i;
typedef __attribute__((ext_vector_type(4)))  int          v4i;

static __device__ __forceinline__ v8f wmma_f16(v16h a, v16h b, v8f c) {
  return __builtin_amdgcn_wmma_f32_16x16x32_f16(false, a, false, b, (short)0, c, false, false);
}

static __device__ __forceinline__ float bperm_f(int byteIdx, float v) {
  return __builtin_bit_cast(float,
      __builtin_amdgcn_ds_bpermute(byteIdx, __builtin_bit_cast(int, v)));
}

// Generic shared pointer -> LDS byte offset (aperture: addr[31:0] is the LDS offset)
static __device__ __forceinline__ uint32_t lds_offset(const void* p) {
  return (uint32_t)(uintptr_t)p;
}

// Issue a TDM 2-D (or 1-D if tile_d1==0) f16 tile load: global -> LDS.
// D# group0: count=1 | lds_addr | global_addr | type=2
// D# group1: data_size=2B, tensor_dim0/1, tile_dim0/1, tensor_dim0_stride
// This toolchain (clang-23 / therock) uses the 6-arg builtin:
//   (uint32x4 g0, int32x8 g1, int32x4 g2, int32x4 g3, int32x8 extra, i32 cpol)
static __device__ __forceinline__ void tdm_load_f16(
    uint32_t lds_off, const void* gptr,
    uint32_t tensor_d0, uint32_t tensor_d1,
    uint32_t tile_d0, uint32_t tile_d1, uint64_t stride0)
{
  const uint64_t ga = (uint64_t)(uintptr_t)gptr;
  v4u g0;
  g0[0] = 1u;                                                  // count=1, user descriptor
  g0[1] = lds_off;                                             // lds_addr (bytes)
  g0[2] = (uint32_t)ga;                                        // global_addr[31:0]
  g0[3] = (uint32_t)((ga >> 32) & 0x01FFFFFFu) | (2u << 30);   // addr[56:32], type=2
  v8i g1;
  g1[0] = (int)(1u << 16);                                     // data_size=1 -> 2 bytes
  g1[1] = (int)((tensor_d0 & 0xFFFFu) << 16);                  // tensor_dim0[15:0]
  g1[2] = (int)((tensor_d0 >> 16) | ((tensor_d1 & 0xFFFFu) << 16));
  g1[3] = (int)((tensor_d1 >> 16) | (tile_d0 << 16));          // tile_dim0
  g1[4] = (int)(tile_d1 & 0xFFFFu);                            // tile_dim1 (0 = 1-D), tile_dim2=0
  g1[5] = (int)(uint32_t)(stride0 & 0xFFFFFFFFu);              // tensor_dim0_stride[31:0]
  g1[6] = (int)(uint32_t)((stride0 >> 32) & 0xFFFFu);          // stride[47:32]; stride1=0
  g1[7] = 0;
  const v4i z4 = {0, 0, 0, 0};                                 // groups 2/3 unused (<=2D)
  const v8i z8 = {0, 0, 0, 0, 0, 0, 0, 0};
  __builtin_amdgcn_tensor_load_to_lds(g0, g1, z4, z4, z8, 0);
}

// A-layout loader for v_wmma_f32_16x16x32_f16 (16x32 f16 A tile, contiguous k order):
//  lanes 0-15  : halves0-15 = K k0..k0+15
//  lanes 16-31 : halves0-15 = K k0+16..k0+31
static __device__ __forceinline__ v16h loadA(const _Float16* rowBase, int k0, bool hi) {
  const _Float16* p = rowBase + k0 + (hi ? 16 : 0);
  v8h lo = *(const v8h*)(p);
  v8h up = *(const v8h*)(p + 8);
  return __builtin_shufflevector(lo, up, 0,1,2,3,4,5,6,7,8,9,10,11,12,13,14,15);
}

// ---------------------------------------------------------------------------
// Kernel 1: fused QKV projection (one wave = 16 s-rows of one (b,h)).
// Q pre-scaled by 0.125*log2(e); V stored transposed [B,H,E,S].
// ---------------------------------------------------------------------------
static __device__ __forceinline__ void proj_mat(
    const _Float16* __restrict__ x, const _Float16* __restrict__ W,
    const _Float16* __restrict__ bvec, _Float16* __restrict__ out,
    bool storeT, float scale, long inRow, int bh, int s0, int l, bool hi)
{
  const _Float16* xr = x + inRow;
  v16h a0 = loadA(xr, 0, hi);
  v16h a1 = loadA(xr, 32, hi);
#pragma unroll
  for (int t = 0; t < 4; ++t) {
    const int n0 = t * 16;
    v16h b0 = *(const v16h*)(W + (n0 + l) * 64 + (hi ? 16 : 0));
    v16h b1 = *(const v16h*)(W + (n0 + l) * 64 + 32 + (hi ? 16 : 0));
    v8f c = {};
    c = wmma_f16(a0, b0, c);
    c = wmma_f16(a1, b1, c);
    const float bias = (float)bvec[n0 + l];
#pragma unroll
    for (int r = 0; r < 8; ++r) {
      const int row = s0 + r + (hi ? 8 : 0);
      const _Float16 val = (_Float16)((c[r] + bias) * scale);
      if (storeT) out[((long)bh * E_ + n0 + l) * S_ + row] = val;   // V^T [BH,E,S]
      else        out[((long)bh * S_ + row) * E_ + n0 + l] = val;   // [BH,S,E]
    }
  }
}

__global__ __launch_bounds__(256) void qkv_proj_kernel(
    const _Float16* __restrict__ q_in, const _Float16* __restrict__ k_in,
    const _Float16* __restrict__ v_in,
    const _Float16* __restrict__ Wq, const _Float16* __restrict__ bq,
    const _Float16* __restrict__ Wk, const _Float16* __restrict__ bk,
    const _Float16* __restrict__ Wv, const _Float16* __restrict__ bv,
    _Float16* __restrict__ q_ws, _Float16* __restrict__ k_ws,
    _Float16* __restrict__ vt_ws)
{
  const int w    = (blockIdx.x * 256 + threadIdx.x) >> 5;
  const int lane = threadIdx.x & 31;
  const int l    = lane & 15;
  const bool hi  = (lane & 16) != 0;
  const int bh   = w >> 7;
  const int b    = bh >> 3, h = bh & 7;
  const int s0   = (w & 127) << 4;
  const long inRow = ((long)(b * S_ + s0 + l) * H_ + h) * E_;
  const float QSCALE = 0.125f * 1.44269504088896f;   // 1/sqrt(E) * log2(e)
  proj_mat(q_in, Wq, bq, q_ws,  false, QSCALE, inRow, bh, s0, l, hi);
  proj_mat(k_in, Wk, bk, k_ws,  false, 1.0f,   inRow, bh, s0, l, hi);
  proj_mat(v_in, Wv, bv, vt_ws, true,  1.0f,   inRow, bh, s0, l, hi);
}

// ---------------------------------------------------------------------------
// Kernel 2: block-cooperative flash attention.
// One block (8 waves) owns a 128-row Q band of one (b,h). K/V chunks are
// staged global->LDS by the Tensor Data Mover (double buffered, TENSORcnt),
// so each chunk is fetched once per block instead of once per wave.
// ---------------------------------------------------------------------------
__global__ __launch_bounds__(256) void attn_kernel(
    const _Float16* __restrict__ Q, const _Float16* __restrict__ K,
    const _Float16* __restrict__ VT, _Float16* __restrict__ out)
{
  __shared__ _Float16 kbuf[2][CHUNK * E_];   // 2 x 8KB : K rows kv0..kv0+63 (row-major, contiguous)
  __shared__ _Float16 vbuf[2][E_ * CHUNK];   // 2 x 8KB : V^T rows e=0..63, cols kv0..kv0+63
  __shared__ float    sscr[8][16 * 32];      // 16KB    : per-wave score transpose scratch

  const int wi   = threadIdx.x >> 5;
  const int lane = threadIdx.x & 31;
  const int l    = lane & 15;
  const bool hi  = (lane & 16) != 0;
  const int bh   = blockIdx.x >> 4;          // 16 (b,h) pairs
  const int band = blockIdx.x & 15;          // 16 bands of 128 rows
  const int s0   = band * 128 + wi * 16;

  const _Float16* Qb = Q  + ((long)bh * S_ + s0) * E_;
  const _Float16* Kb = K  + (long)bh * S_ * E_;
  const _Float16* Vb = VT + (long)bh * E_ * S_;

  v16h qa0 = loadA(Qb + l * E_, 0, hi);
  v16h qa1 = loadA(Qb + l * E_, 32, hi);

  v8f o0 = {}, o1 = {}, o2 = {}, o3 = {};
  float m = -__builtin_inff(), lsum = 0.0f;
  const int partner = (lane ^ 16) << 2;
  const int bcastHi = hi ? 32 : 0;
  float* myScr = sscr[wi];

  // prologue: stage chunk 0 into buffer 0
  if (wi == 0) {
    tdm_load_f16(lds_offset(&kbuf[0][0]), Kb, CHUNK * E_, 1, CHUNK * E_, 0, CHUNK * E_);
    tdm_load_f16(lds_offset(&vbuf[0][0]), Vb, S_, E_, CHUNK, E_, S_);
  }
  __builtin_amdgcn_s_wait_tensorcnt(0);
  __syncthreads();

  for (int c = 0; c < NCHUNK; ++c) {
    const int buf = c & 1;
    if (wi == 0 && c + 1 < NCHUNK) {   // prefetch next chunk into the other buffer
      const int kv1 = (c + 1) * CHUNK;
      tdm_load_f16(lds_offset(&kbuf[buf ^ 1][0]), Kb + (long)kv1 * E_,
                   CHUNK * E_, 1, CHUNK * E_, 0, CHUNK * E_);
      tdm_load_f16(lds_offset(&vbuf[buf ^ 1][0]), Vb + kv1,
                   S_, E_, CHUNK, E_, S_);
    }
    const _Float16* kl = kbuf[buf];
    const _Float16* vl = vbuf[buf];

#pragma unroll
    for (int sub = 0; sub < CHUNK; sub += 32) {
      // scores: B[k][n] = K[kv0+sub+n][k]; lane n holds a contiguous 32B K-row slice (LDS)
      const _Float16* kr0 = kl + (sub + l) * E_ + (hi ? 16 : 0);
      const _Float16* kr1 = kl + (sub + 16 + l) * E_ + (hi ? 16 : 0);
      v8f c0 = {}, c1 = {};
      c0 = wmma_f16(qa0, *(const v16h*)(kr0), c0);
      c0 = wmma_f16(qa1, *(const v16h*)(kr0 + 32), c0);
      c1 = wmma_f16(qa0, *(const v16h*)(kr1), c1);
      c1 = wmma_f16(qa1, *(const v16h*)(kr1 + 32), c1);

      // C-layout -> LDS [row][col] (16x32 f32), then row-per-lane gather
#pragma unroll
      for (int r = 0; r < 8; ++r) {
        const int row = r + (hi ? 8 : 0);
        myScr[row * 32 + l]      = c0[r];
        myScr[row * 32 + 16 + l] = c1[r];
      }
      __builtin_amdgcn_wave_barrier();   // same-wave DS is in-order; fence compiler motion

      float s[16];
      const int base = l * 32 + (hi ? 16 : 0);
#pragma unroll
      for (int i = 0; i < 8; ++i) { s[i] = myScr[base + i]; s[8 + i] = myScr[base + 8 + i]; }

      float cmax = s[0];
#pragma unroll
      for (int i = 1; i < 16; ++i) cmax = fmaxf(cmax, s[i]);
      cmax = fmaxf(cmax, bperm_f(partner, cmax));
      const float mnew = fmaxf(m, cmax);
      const float corr = __builtin_amdgcn_exp2f(m - mnew);   // exp2 domain

      float ps = 0.0f;
      v16h pa;
#pragma unroll
      for (int i = 0; i < 16; ++i) {
        const float p = __builtin_amdgcn_exp2f(s[i] - mnew);
        ps += p;
        pa[i] = (_Float16)p;                                 // already A-layout
      }
      ps += bperm_f(partner, ps);
      lsum = lsum * corr + ps;
      m = mnew;

#pragma unroll
      for (int r = 0; r < 8; ++r) {
        const float cr = bperm_f(4 * r + bcastHi, corr);
        o0[r] *= cr; o1[r] *= cr; o2[r] *= cr; o3[r] *= cr;
      }

      // PV: B[k][n] = VT[e0+n][kv0+sub+k]; lane n reads 32B of V^T row e0+n from LDS
      const _Float16* vc = vl + l * CHUNK + sub + (hi ? 16 : 0);
      o0 = wmma_f16(pa, *(const v16h*)(vc),              o0);
      o1 = wmma_f16(pa, *(const v16h*)(vc + 16 * CHUNK), o1);
      o2 = wmma_f16(pa, *(const v16h*)(vc + 32 * CHUNK), o2);
      o3 = wmma_f16(pa, *(const v16h*)(vc + 48 * CHUNK), o3);
    }

    // publish next buffer: wave0 waits its TDM, then barrier releases everyone
    __builtin_amdgcn_s_wait_tensorcnt(0);
    __syncthreads();
  }

  const float linv = 1.0f / lsum;
  _Float16* ob = out + ((long)bh * S_ + s0) * E_;
#pragma unroll
  for (int r = 0; r < 8; ++r) {
    const float li = bperm_f(4 * r + bcastHi, linv);
    const int row = r + (hi ? 8 : 0);
    _Float16* orow = ob + row * E_ + l;
    orow[0]  = (_Float16)(o0[r] * li);
    orow[16] = (_Float16)(o1[r] * li);
    orow[32] = (_Float16)(o2[r] * li);
    orow[48] = (_Float16)(o3[r] * li);
  }
}

extern "C" void kernel_launch(void* const* d_in, const int* in_sizes, int n_in,
                              void* d_out, int out_size, void* d_ws, size_t ws_size,
                              hipStream_t stream) {
  const _Float16* q_in = (const _Float16*)d_in[0];
  const _Float16* k_in = (const _Float16*)d_in[1];
  const _Float16* v_in = (const _Float16*)d_in[2];
  const _Float16* Wq   = (const _Float16*)d_in[3];
  const _Float16* bq   = (const _Float16*)d_in[4];
  const _Float16* Wk   = (const _Float16*)d_in[5];
  const _Float16* bk   = (const _Float16*)d_in[6];
  const _Float16* Wv   = (const _Float16*)d_in[7];
  const _Float16* bv   = (const _Float16*)d_in[8];

  _Float16* q_ws  = (_Float16*)d_ws;
  _Float16* k_ws  = q_ws + (size_t)BH_ * S_ * E_;
  _Float16* vt_ws = k_ws + (size_t)BH_ * S_ * E_;
  _Float16* outp  = (_Float16*)d_out;

  qkv_proj_kernel<<<256, 256, 0, stream>>>(q_in, k_in, v_in, Wq, bq, Wk, bk,
                                           Wv, bv, q_ws, k_ws, vt_ws);
  attn_kernel<<<256, 256, 0, stream>>>(q_ws, k_ws, vt_ws, outp);
}
